// EarthSpecificBlock_9698036154399
// MI455X (gfx1250) — matrix operand order repaired and checked by reading, not compiled, not run on hardware
//
#include <hip/hip_runtime.h>
#include <hip/hip_bf16.h>

// ---------------------------------------------------------------------------
// EarthSpecificBlock (Pangu 3D-Swin attention block), MI455X / gfx1250.
// bf16 WMMA (v_wmma_f32_16x16x32_bf16) everywhere, f32 accumulate.
// Window = (2,6,12) -> N=144 tokens, C=192, 6 heads x hd=32.
// nZ=4, nH=16, nW=15 -> Bw=960 windows. type = window % 64. f = 15.
// ---------------------------------------------------------------------------

typedef __attribute__((ext_vector_type(16))) __bf16 v16bf;
typedef __attribute__((ext_vector_type(8)))  float  v8f;

union ABu { v16bf v; unsigned int u[8]; };

// A-matrix 16x32 bf16 fragment (M x K), source row-major, stride in elements.
// Lane L(0..15): M=L, halves: e0..7 -> K=2p+8*hi(grp0), e8..15 -> K=2p+8+8*hi.
__device__ __forceinline__ v16bf load_a_frag(const __bf16* p, int stride) {
  const int lane = threadIdx.x & 31;
  const int ml = lane & 15, hi = lane >> 4;
  const __bf16* row = p + ml * stride + 8 * hi;
  ABu r;
#pragma unroll
  for (int q = 0; q < 8; ++q) {
    int k = 2 * q + (q >= 4 ? 8 : 0);
    r.u[q] = *reinterpret_cast<const unsigned int*>(row + k);
  }
  return r.v;
}

// B-matrix 32x16 bf16 fragment (K x N); memory holds value(k,n) at p[n*stride+k]
// (i.e. "n-major / k-contiguous"). Lane L: N=L%16, e0..7 -> K=8*hi+e, e8..15 -> K=16+8*hi+(e-8).
__device__ __forceinline__ v16bf load_b_frag(const __bf16* p, int stride) {
  const int lane = threadIdx.x & 31;
  const int nl = lane & 15, hi = lane >> 4;
  const __bf16* row = p + nl * stride + 8 * hi;
  ABu r;
#pragma unroll
  for (int q = 0; q < 8; ++q)
    r.u[q] = *reinterpret_cast<const unsigned int*>(row + 2 * q + (q >= 4 ? 8 : 0));
  return r.v;
}

__device__ __forceinline__ v8f wmma_bf16(v16bf a, v16bf b, v8f c) {
  return __builtin_amdgcn_wmma_f32_16x16x32_bf16(
      /*neg_a=*/false, a, /*neg_b=*/false, b,
      /*c_mod=*/(short)0, c, /*reuse_a=*/false, /*reuse_b=*/false);
}

// ---------------------------------------------------------------------------
// K1: convert+transpose weights to bf16 in workspace.
//   ws[0      .. 110592) : wqkvT  (576 x 192)   T[n][k] = w_qkv[k*576+n]
//   ws[110592 .. 147456) : wprojT (192 x 192)   T[n][k] = w_proj[k*192+n]
//   ws[147456 .. 184320) : wmlpT  (192 x 192)
// ---------------------------------------------------------------------------
__global__ void __launch_bounds__(256) prep_weights(
    const float* __restrict__ wqkv, const float* __restrict__ wproj,
    const float* __restrict__ wmlp, __bf16* __restrict__ ws) {
  int i = blockIdx.x * 256 + threadIdx.x;
  if (i < 110592) {
    int n = i / 192, k = i - n * 192;
    ws[i] = (__bf16)wqkv[k * 576 + n];
  } else if (i < 147456) {
    int j = i - 110592; int n = j / 192, k = j - n * 192;
    ws[i] = (__bf16)wproj[k * 192 + n];
  } else if (i < 184320) {
    int j = i - 147456; int n = j / 192, k = j - n * 192;
    ws[i] = (__bf16)wmlp[k * 192 + n];
  }
}

// ---------------------------------------------------------------------------
// K2: fused window-attention block. One workgroup (256 thr = 8 waves) per
// window. Dynamic LDS layout (bytes), total 319488 <= 320KB WGP LDS:
//   [0      , 55296 ) Qbf  144x192 bf16  (pre-scaled by 1/sqrt(32))
//   [55296  , 110592) Kbf  144x192 bf16
//   [110592 , 172032) Vt   192x160 bf16  (row = h*32+d, col = tok, padded)
//   [172032 , 264192) Xbf 144x192 bf16 | S 144x160 f32 | P 144x(320) bf16
//   [264192 , 319488) OB   144x192 bf16  (attention output, proj input)
//   proj-out f32 144x192 overlays [0,110592) (Q,K dead by then)
// ---------------------------------------------------------------------------
__global__ void __launch_bounds__(256) attn_block_kernel(
    const float* __restrict__ x, const float* __restrict__ bqkv,
    const float* __restrict__ bproj, const float* __restrict__ bias_tab,
    const float* __restrict__ g1, const float* __restrict__ b1,
    const __bf16* __restrict__ wqkvT, const __bf16* __restrict__ wprojT,
    float* __restrict__ out) {
  extern __shared__ char smem[];
  __bf16* Qbf = (__bf16*)(smem + 0);
  __bf16* Kbf = (__bf16*)(smem + 55296);
  __bf16* Vt  = (__bf16*)(smem + 110592);
  __bf16* Xbf = (__bf16*)(smem + 172032);
  float*  S   = (float*)(smem + 172032);
  __bf16* Pbf = (__bf16*)(smem + 172032);   // row stride 320 bf16 (=640B)
  __bf16* OB  = (__bf16*)(smem + 264192);
  float*  PO  = (float*)(smem + 0);         // overlays Q,K

  const int b = blockIdx.x;
  const int tid = threadIdx.x;
  // Wave id is uniform by construction; readfirstlane makes it an SGPR so
  // tile-loop induction and region branches are scalar (no EXEC masking).
  const int wave = __builtin_amdgcn_readfirstlane(tid >> 5);
  const int zi = b / 240, rem = b % 240, hwi = rem / 15, wwi = rem % 15;
  const int type = b & 63;                  // window % 64 (f=15 broadcast)

  // ---- Phase A: gather window tokens, convert to bf16; zero Vt K-padding ---
  for (int ii = tid; ii < 144 * 192; ii += 256) {
    int t = ii / 192, c = ii - t * 192;
    int tz = t / 72, tr = t - tz * 72, th = tr / 12, tw = tr - th * 12;
    int g = ((zi * 2 + tz) * 96 + (hwi * 6 + th)) * 180 + (wwi * 12 + tw);
    Xbf[ii] = (__bf16)x[(size_t)g * 192 + c];
  }
  for (int ii = tid; ii < 192 * 16; ii += 256) {
    int rr = ii >> 4, tk = 144 + (ii & 15);
    Vt[rr * 160 + tk] = (__bf16)0.0f;       // P pads are 0, but keep B finite
  }
  __syncthreads();

  const int lane = tid & 31, nl = lane & 15, hig = lane >> 4;

  // ---- Phase B: QKV = X(144x192) @ Wqkv(192x576) + b ----------------------
  // Region (Q / K / V) is uniform per tile: branch on scalar nt, so the LDS
  // scatter stores need no EXEC masking.
  for (int tt = wave; tt < 9 * 36; tt += 8) {
    const int mt = tt / 36, nt = tt - mt * 36;   // scalar
    v8f acc = {};
#pragma unroll
    for (int ks = 0; ks < 6; ++ks) {
      v16bf a  = load_a_frag(Xbf + (mt * 16) * 192 + ks * 32, 192);
      v16bf bb = load_b_frag(wqkvT + (nt * 16) * 192 + ks * 32, 192);
      acc = wmma_bf16(a, bb, acc);
    }
    const int col = nt * 16 + nl;              // global qkv column 0..575
    const float bias = bqkv[col];
    if (nt < 12) {                             // ---- Q (scaled) ----
#pragma unroll
      for (int r = 0; r < 8; ++r) {
        int tok = mt * 16 + r + 8 * hig;
        Qbf[tok * 192 + col] =
            (__bf16)((acc[r] + bias) * 0.17677669529663689f);  // 1/sqrt(32)
      }
    } else if (nt < 24) {                      // ---- K ----
#pragma unroll
      for (int r = 0; r < 8; ++r) {
        int tok = mt * 16 + r + 8 * hig;
        Kbf[tok * 192 + (col - 192)] = (__bf16)(acc[r] + bias);
      }
    } else {                                   // ---- V (transposed) ----
      int cv = col - 384;                      // h*32+d
#pragma unroll
      for (int r = 0; r < 8; ++r) {
        int tok = mt * 16 + r + 8 * hig;
        Vt[cv * 160 + tok] = (__bf16)(acc[r] + bias);
      }
    }
  }
  __syncthreads();

  // ---- Phase C: per-head attention ---------------------------------------
  for (int h = 0; h < 6; ++h) {
    // C1: S = Q·Kᵀ + earth-specific bias   (single K=32 WMMA per tile)
    for (int tt = wave; tt < 81; tt += 8) {
      const int mt = tt / 9, nt = tt - mt * 9;   // scalar
      v16bf a  = load_a_frag(Qbf + (mt * 16) * 192 + h * 32, 192);
      v16bf bb = load_b_frag(Kbf + (nt * 16) * 192 + h * 32, 192);
      v8f acc = {};
      acc = wmma_bf16(a, bb, acc);
      int j = nt * 16 + nl;
      int jz = j / 72, jr = j - jz * 72, jh = jr / 12, jw = jr - jh * 12;
#pragma unroll
      for (int r = 0; r < 8; ++r) {
        int i = mt * 16 + r + 8 * hig;
        int iz = i / 72, ir = i - iz * 72, ih = ir / 12, iw = ir - ih * 12;
        int idx = (iz + 2 * jz) * 828 + (ih + 6 * jh) * 23 + (iw - jw + 11);
        S[i * 160 + j] = acc[r] + bias_tab[(idx * 64 + type) * 6 + h];
      }
    }
    __syncthreads();
    // C2: row softmax; write bf16 P in place (2-byte strides within f32 row)
    if (tid < 144) {
      float* row = S + tid * 160;
      float mx = -3.0e38f;
      for (int j = 0; j < 144; ++j) mx = fmaxf(mx, row[j]);
      float sum = 0.f;
      for (int j = 0; j < 144; ++j) { float e = __expf(row[j] - mx); row[j] = e; sum += e; }
      float inv = 1.0f / sum;
      __bf16* prow = Pbf + tid * 320;
      for (int j = 0; j < 144; ++j) { float e = row[j]; prow[j] = (__bf16)(e * inv); }
      for (int j = 144; j < 160; ++j) prow[j] = (__bf16)0.0f;   // K pad
    }
    __syncthreads();
    // C3: O = P(144x160) @ V(160x32), pads contribute 0
    for (int tt = wave; tt < 18; tt += 8) {
      const int mt = tt >> 1, nt = tt & 1;       // scalar
      v8f acc = {};
#pragma unroll
      for (int ks = 0; ks < 5; ++ks) {
        v16bf a  = load_a_frag(Pbf + (mt * 16) * 320 + ks * 32, 320);
        v16bf bb = load_b_frag(Vt + (h * 32 + nt * 16) * 160 + ks * 32, 160);
        acc = wmma_bf16(a, bb, acc);
      }
#pragma unroll
      for (int r = 0; r < 8; ++r) {
        int tok = mt * 16 + r + 8 * hig;
        OB[tok * 192 + h * 32 + nt * 16 + nl] = (__bf16)acc[r];
      }
    }
    __syncthreads();
  }

  // ---- Phase D: proj = OB(144x192) @ Wproj(192x192) + b -------------------
  for (int tt = wave; tt < 108; tt += 8) {
    const int mt = tt / 12, nt = tt - mt * 12;   // scalar
    v8f acc = {};
#pragma unroll
    for (int ks = 0; ks < 6; ++ks) {
      v16bf a  = load_a_frag(OB + (mt * 16) * 192 + ks * 32, 192);
      v16bf bb = load_b_frag(wprojT + (nt * 16) * 192 + ks * 32, 192);
      acc = wmma_bf16(a, bb, acc);
    }
#pragma unroll
    for (int r = 0; r < 8; ++r) {
      int tok = mt * 16 + r + 8 * hig;
      int col = nt * 16 + nl;
      PO[tok * 192 + col] = acc[r] + bproj[col];
    }
  }
  __syncthreads();

  // ---- Phase E: out = shortcut + LayerNorm(proj) (window-merge scatter) ---
  if (tid < 144) {
    int t = tid;
    int tz = t / 72, tr = t - tz * 72, th = tr / 12, tw = tr - th * 12;
    int g = ((zi * 2 + tz) * 96 + (hwi * 6 + th)) * 180 + (wwi * 12 + tw);
    const float* pr = PO + t * 192;
    float mean = 0.f;
    for (int c = 0; c < 192; ++c) mean += pr[c];
    mean *= (1.0f / 192.0f);
    float var = 0.f;
    for (int c = 0; c < 192; ++c) { float d = pr[c] - mean; var += d * d; }
    var *= (1.0f / 192.0f);
    float rstd = rsqrtf(var + 1e-5f);
    for (int c = 0; c < 192; ++c) {
      float ln = (pr[c] - mean) * rstd * g1[c] + b1[c];
      out[(size_t)g * 192 + c] = x[(size_t)g * 192 + c] + ln;
    }
  }
}

// ---------------------------------------------------------------------------
// K3: y = y + LayerNorm(y @ Wmlp + b).  In-place on d_out; each workgroup
// owns 144 rows (reads then writes only its own rows). LDS: Ybf 55296 B +
// MO f32 110592 B = 165888 B dynamic.
// ---------------------------------------------------------------------------
__global__ void __launch_bounds__(256) mlp_kernel(
    const float* __restrict__ g2, const float* __restrict__ b2,
    const __bf16* __restrict__ wmlpT, const float* __restrict__ bmlp,
    float* __restrict__ y) {
  extern __shared__ char smem[];
  __bf16* Ybf = (__bf16*)smem;
  float*  MO  = (float*)(smem + 55296);
  const int tid = threadIdx.x;
  const int wave = __builtin_amdgcn_readfirstlane(tid >> 5);
  const int lane = tid & 31, nl = lane & 15, hig = lane >> 4;
  const size_t r0 = (size_t)blockIdx.x * 144;

  for (int ii = tid; ii < 144 * 192; ii += 256)
    Ybf[ii] = (__bf16)y[r0 * 192 + ii];
  __syncthreads();

  for (int tt = wave; tt < 108; tt += 8) {
    const int mt = tt / 12, nt = tt - mt * 12;   // scalar
    v8f acc = {};
#pragma unroll
    for (int ks = 0; ks < 6; ++ks) {
      v16bf a  = load_a_frag(Ybf + (mt * 16) * 192 + ks * 32, 192);
      v16bf bb = load_b_frag(wmlpT + (nt * 16) * 192 + ks * 32, 192);
      acc = wmma_bf16(a, bb, acc);
    }
#pragma unroll
    for (int r = 0; r < 8; ++r) {
      int tok = mt * 16 + r + 8 * hig;
      int col = nt * 16 + nl;
      MO[tok * 192 + col] = acc[r] + bmlp[col];
    }
  }
  __syncthreads();

  if (tid < 144) {
    const float* pr = MO + tid * 192;
    float mean = 0.f;
    for (int c = 0; c < 192; ++c) mean += pr[c];
    mean *= (1.0f / 192.0f);
    float var = 0.f;
    for (int c = 0; c < 192; ++c) { float d = pr[c] - mean; var += d * d; }
    var *= (1.0f / 192.0f);
    float rstd = rsqrtf(var + 1e-5f);
    size_t base = (r0 + tid) * 192;
    for (int c = 0; c < 192; ++c) {
      float ln = (pr[c] - mean) * rstd * g2[c] + b2[c];
      y[base + c] = y[base + c] + ln;
    }
  }
}

// ---------------------------------------------------------------------------
extern "C" void kernel_launch(void* const* d_in, const int* in_sizes, int n_in,
                              void* d_out, int out_size, void* d_ws, size_t ws_size,
                              hipStream_t stream) {
  (void)in_sizes; (void)n_in; (void)out_size; (void)ws_size;
  const float* x          = (const float*)d_in[0];
  const float* w_qkv      = (const float*)d_in[1];
  const float* b_qkv      = (const float*)d_in[2];
  const float* w_proj     = (const float*)d_in[3];
  const float* b_proj     = (const float*)d_in[4];
  const float* bias_table = (const float*)d_in[5];
  const float* g1         = (const float*)d_in[6];
  const float* b1         = (const float*)d_in[7];
  const float* g2         = (const float*)d_in[8];
  const float* b2         = (const float*)d_in[9];
  const float* w_mlp      = (const float*)d_in[10];
  const float* b_mlp      = (const float*)d_in[11];
  float* out = (float*)d_out;
  __bf16* wsb = (__bf16*)d_ws;

  prep_weights<<<720, 256, 0, stream>>>(w_qkv, w_proj, w_mlp, wsb);
  attn_block_kernel<<<960, 256, 319488, stream>>>(
      x, b_qkv, b_proj, bias_table, g1, b1, wsb, wsb + 110592, out);
  mlp_kernel<<<960, 256, 165888, stream>>>(g2, b2, wsb + 147456, b_mlp, out);
}
